// GNNClassifier_61503931679088
// MI455X (gfx1250) — matrix operand order repaired and checked by reading, not compiled, not run on hardware
//
#include <hip/hip_runtime.h>
#include <hip/hip_bf16.h>
#include <math.h>
#include <stdint.h>

// GGNN (GatedGraphConv + GRUCell + ELU head) for MI455X / gfx1250.
//
//  * GEMMs: v_wmma_f32_16x16x32_bf16, f32 accumulate.
//  * Weights pre-packed ONCE to bf16 in WMMA B-fragment order (one 32B
//    load per lane per fragment, no per-block conversion).
//  * Activation slabs staged per block in LDS as bf16 in A-fragment order
//    (1x global read instead of 6x, waves fetch via ds_load_b128).
//  * Edge aggregation: CSR-by-destination built once per launch
//    (histogram + hierarchical scan + fill), then per-node gather-sum
//    in exact fp32 -- zero fp atomics, coalesced writes.

#define NN   50000          // nodes  (== 3125 * 16)
#define DD   96             // feature dim
#define EE   800000         // edges  (== 3125 * 256)
#define CC   10             // classes
#define ROWB (NN / 16)      // 3125 row-blocks of 16
#define NB   ((NN + 255) / 256)   // 196 scan blocks

typedef __attribute__((ext_vector_type(16))) __bf16 v16bf;
typedef __attribute__((ext_vector_type(8)))  float  v8f;

#define WMMA_BF16(A, B, C) \
  __builtin_amdgcn_wmma_f32_16x16x32_bf16(false, (A), false, (B), (short)0, (C), false, false)

// ---------------------------------------------------------------------------
// Fragment layouts used here (wave32; contiguous k order, not the ISA table order):
//  A (16x32 MxK): lane = 16*half + M ;  elem i -> K=16*half+i
//  B (32x16 KxN): lane = 16*half + N ;  elem i -> K=16*half+i
// ---------------------------------------------------------------------------

// Pack one weight matrix W[outdim][96] (row-major fp32) into bf16 B-fragments:
// frag f = tile*3 + kc ; out[(f*32 + lane)*16 + i] = W[tile*16 + lane%16][kc*32 + 16*(lane/16) + i]
__global__ __launch_bounds__(512) void pack_w_kernel(const float* __restrict__ W,
                                                     __bf16* __restrict__ out) {
  const int f    = blockIdx.x;            // tile*3 + kc
  const int tile = f / 3, kc = f % 3;
  const int lane = threadIdx.x >> 4;      // 0..31
  const int i    = threadIdx.x & 15;
  const int n    = tile * 16 + (lane & 15);
  const int k    = kc * 32 + (lane >> 4) * 16 + i;
  out[((size_t)f * 32 + lane) * 16 + i] = (__bf16)W[(size_t)n * DD + k];
}

// ---------------------------------------------------------------------------
// CSR build: counts -> exclusive scan -> fill
// ---------------------------------------------------------------------------
__global__ void zero_int_kernel(int* __restrict__ p, int n) {
  const int i = blockIdx.x * blockDim.x + threadIdx.x;
  if (i < n) p[i] = 0;
}

__global__ void count_kernel(const int* __restrict__ dst, int* __restrict__ cnt) {
  const int e = blockIdx.x * 256 + threadIdx.x;     // EE == 3125*256
  atomicAdd(&cnt[dst[e]], 1);
}

__global__ __launch_bounds__(256) void scan_block_kernel(const int* __restrict__ cnt,
                                                         int* __restrict__ excl,
                                                         int* __restrict__ bsums) {
  __shared__ int s[256];
  const int t   = threadIdx.x;
  const int idx = blockIdx.x * 256 + t;
  const int v   = (idx < NN) ? cnt[idx] : 0;
  s[t] = v;
  __syncthreads();
  int x = v;
#pragma unroll
  for (int off = 1; off < 256; off <<= 1) {
    const int y = (t >= off) ? s[t - off] : 0;
    __syncthreads();
    x += y;
    s[t] = x;
    __syncthreads();
  }
  if (idx < NN) excl[idx] = x - v;                  // exclusive prefix
  if (t == 255) bsums[blockIdx.x] = x;              // block total
}

__global__ __launch_bounds__(256) void scan_sums_kernel(int* __restrict__ bsums, int nb) {
  __shared__ int s[256];
  const int t = threadIdx.x;
  const int v = (t < nb) ? bsums[t] : 0;
  s[t] = v;
  __syncthreads();
  int x = v;
#pragma unroll
  for (int off = 1; off < 256; off <<= 1) {
    const int y = (t >= off) ? s[t - off] : 0;
    __syncthreads();
    x += y;
    s[t] = x;
    __syncthreads();
  }
  if (t < nb) bsums[t] = x - v;                     // exclusive block offsets
}

__global__ void add_off_kernel(int* __restrict__ rs, const int* __restrict__ bsums) {
  const int idx = blockIdx.x * 256 + threadIdx.x;
  if (idx < NN) rs[idx] += bsums[blockIdx.x];
}

__global__ void set_last_kernel(int* __restrict__ rs) { rs[NN] = EE; }

__global__ void fill_csr_kernel(const int* __restrict__ src, const int* __restrict__ dst,
                                const int* __restrict__ rs, int* __restrict__ fill,
                                int* __restrict__ csr) {
  const int e = blockIdx.x * 256 + threadIdx.x;
  const int d = dst[e];
  const int pos = rs[d] + atomicAdd(&fill[d], 1);
  csr[pos] = src[e];
}

// ---------------------------------------------------------------------------
// LDS staging: 16 rows x 96 fp32 -> bf16 in A-fragment order (3 kc-chunks).
// 192 threads x 8 contiguous elems; bijection (row,k) <-> (kc,lane,i).
// ---------------------------------------------------------------------------
__device__ __forceinline__ void stage_a_slab(const float* __restrict__ g, int row0,
                                             __bf16* __restrict__ lds, int tid) {
  const int base = tid * 8;                  // 192*8 == 16*96
  const int r    = base / DD;
  const int k0   = base % DD;                // 8 elems never cross a row (96 % 8 == 0)
  const float* src = g + (size_t)(row0 + r) * DD + k0;
#pragma unroll
  for (int j = 0; j < 8; ++j) {
    const int k    = k0 + j;
    const int kc   = k >> 5;
    const int q    = k & 31;
    const int half = q >> 4;
    const int i    = q & 15;
    const int lane = half * 16 + r;
    lds[(kc * 32 + lane) * 16 + i] = (__bf16)src[j];
  }
}

// ---------------------------------------------------------------------------
// m = h @ W_e^T + b_e      (ROWB blocks x 192 threads = 6 waves)
// ---------------------------------------------------------------------------
__global__ __launch_bounds__(192) void msg_kernel(const float* __restrict__ h,
                                                  const __bf16* __restrict__ PWe,
                                                  const float* __restrict__ be,
                                                  float* __restrict__ m) {
  __shared__ v16bf lsA[3 * 32];
  const int tid  = threadIdx.x;
  const int row0 = blockIdx.x * 16;

  stage_a_slab(h, row0, (__bf16*)lsA, tid);
  __syncthreads();

  const int wave = tid >> 5;
  const int lane = tid & 31;
  const int r    = lane & 15;
  const int half = lane >> 4;
  const v16bf* PB = (const v16bf*)PWe;

  v8f acc = {};
#pragma unroll
  for (int kc = 0; kc < 3; ++kc) {
    const v16bf a = lsA[kc * 32 + lane];
    const v16bf b = PB[((size_t)wave * 3 + kc) * 32 + lane];
    acc = WMMA_BF16(a, b, acc);
  }

  const int   col  = wave * 16 + r;
  const float bias = be[col];
#pragma unroll
  for (int v = 0; v < 8; ++v) {
    const int row = row0 + half * 8 + v;
    m[(size_t)row * DD + col] = acc[v] + bias;
  }
}

// ---------------------------------------------------------------------------
// a[n][:] = sum over in-edges of m[src][:]   (CSR gather-sum, no atomics)
// ---------------------------------------------------------------------------
__global__ __launch_bounds__(96) void gather_kernel(const float* __restrict__ m,
                                                    const int* __restrict__ rs,
                                                    const int* __restrict__ csr,
                                                    float* __restrict__ a) {
  const int n   = blockIdx.x;
  const int c   = threadIdx.x;
  const int beg = rs[n], end = rs[n + 1];
  float acc = 0.f;
  for (int i = beg; i < end; ++i) {
    const int s = csr[i];                    // uniform -> scalar load
    acc += m[(size_t)s * DD + c];            // coalesced 384B row read
  }
  a[(size_t)n * DD + c] = acc;
}

// ---------------------------------------------------------------------------
// Fused GRU cell (18 WMMAs per wave: r/z/n gates for both a@W_ih^T, h@W_hh^T)
// ---------------------------------------------------------------------------
__global__ __launch_bounds__(192) void gru_kernel(const float* __restrict__ a,
                                                  const float* __restrict__ h,
                                                  const __bf16* __restrict__ PWih,
                                                  const __bf16* __restrict__ PWhh,
                                                  const float* __restrict__ bih,
                                                  const float* __restrict__ bhh,
                                                  float* __restrict__ h_out) {
  __shared__ v16bf lsA[3 * 32];
  __shared__ v16bf lsH[3 * 32];
  const int tid  = threadIdx.x;
  const int row0 = blockIdx.x * 16;

  stage_a_slab(a, row0, (__bf16*)lsA, tid);
  stage_a_slab(h, row0, (__bf16*)lsH, tid);
  __syncthreads();

  const int wave = tid >> 5;
  const int lane = tid & 31;
  const int r    = lane & 15;
  const int half = lane >> 4;
  const v16bf* PI = (const v16bf*)PWih;
  const v16bf* PH = (const v16bf*)PWhh;

  v8f xr = {}, xz = {}, xn = {}, hr = {}, hz = {}, hg = {};
#pragma unroll
  for (int kc = 0; kc < 3; ++kc) {
    const v16bf af = lsA[kc * 32 + lane];
    const v16bf hf = lsH[kc * 32 + lane];
    // W tile for gate g, column block `wave` is tile (6g + wave); frag = tile*3 + kc
    v16bf b;
    b = PI[((size_t)(0 * 6 + wave) * 3 + kc) * 32 + lane];  xr = WMMA_BF16(af, b, xr);
    b = PI[((size_t)(1 * 6 + wave) * 3 + kc) * 32 + lane];  xz = WMMA_BF16(af, b, xz);
    b = PI[((size_t)(2 * 6 + wave) * 3 + kc) * 32 + lane];  xn = WMMA_BF16(af, b, xn);
    b = PH[((size_t)(0 * 6 + wave) * 3 + kc) * 32 + lane];  hr = WMMA_BF16(hf, b, hr);
    b = PH[((size_t)(1 * 6 + wave) * 3 + kc) * 32 + lane];  hz = WMMA_BF16(hf, b, hz);
    b = PH[((size_t)(2 * 6 + wave) * 3 + kc) * 32 + lane];  hg = WMMA_BF16(hf, b, hg);
  }

  const int   col = wave * 16 + r;
  const float bir = bih[col], biz = bih[DD + col], bin = bih[2 * DD + col];
  const float bhr = bhh[col], bhz = bhh[DD + col], bhn = bhh[2 * DD + col];

#pragma unroll
  for (int v = 0; v < 8; ++v) {
    const int   row = row0 + half * 8 + v;
    const float hv  = h[(size_t)row * DD + col];
    const float rg  = 1.f / (1.f + __expf(-((xr[v] + bir) + (hr[v] + bhr))));
    const float zg  = 1.f / (1.f + __expf(-((xz[v] + biz) + (hz[v] + bhz))));
    const float ng  = tanhf((xn[v] + bin) + rg * (hg[v] + bhn));
    h_out[(size_t)row * DD + col] = (1.f - zg) * ng + zg * hv;
  }
}

// ---------------------------------------------------------------------------
// logits = elu(h) @ W_fc^T + b_fc   (one wave per node, shuffle reduction)
// ---------------------------------------------------------------------------
__device__ __forceinline__ float elu1(float x) { return x > 0.f ? x : (__expf(x) - 1.f); }

__global__ __launch_bounds__(128) void head_kernel(const float* __restrict__ h,
                                                   const float* __restrict__ Wfc,
                                                   const float* __restrict__ bfc,
                                                   float* __restrict__ out) {
  const int node = blockIdx.x * 4 + (threadIdx.x >> 5);
  const int lane = threadIdx.x & 31;
  const float* hp = h + (size_t)node * DD;
  const float e0 = elu1(hp[lane]);
  const float e1 = elu1(hp[lane + 32]);
  const float e2 = elu1(hp[lane + 64]);
#pragma unroll
  for (int c = 0; c < CC; ++c) {
    const float* w = Wfc + c * DD;
    float p = e0 * w[lane] + e1 * w[lane + 32] + e2 * w[lane + 64];
#pragma unroll
    for (int off = 16; off > 0; off >>= 1) p += __shfl_down(p, off, 32);
    if (lane == 0) out[node * CC + c] = p + bfc[c];
  }
}

// ---------------------------------------------------------------------------
extern "C" void kernel_launch(void* const* d_in, const int* in_sizes, int n_in,
                              void* d_out, int out_size, void* d_ws, size_t ws_size,
                              hipStream_t stream) {
  const float* x    = (const float*)d_in[0];
  const int*   src  = (const int*)  d_in[1];
  const int*   dst  = (const int*)  d_in[2];
  const float* W_e  = (const float*)d_in[3];
  const float* b_e  = (const float*)d_in[4];
  const float* W_ih = (const float*)d_in[5];
  const float* W_hh = (const float*)d_in[6];
  const float* b_ih = (const float*)d_in[7];
  const float* b_hh = (const float*)d_in[8];
  const float* W_fc = (const float*)d_in[9];
  const float* b_fc = (const float*)d_in[10];
  (void)in_sizes; (void)n_in; (void)out_size; (void)ws_size;

  // ---- workspace layout (~61.5 MB) ----
  const size_t ND = (size_t)NN * DD;
  float* P = (float*)d_ws;                      // m / h ping
  float* Q = P + ND;                            // aggregated messages 'a'
  float* R = Q + ND;                            // h pong
  int* row_start = (int*)(R + ND);              // NN+1
  int* fill      = row_start + (NN + 1);        // NN  (counts, then fill cursor)
  int* bsums     = fill + NN;                   // 256
  int* csr       = bsums + 256;                 // EE
  uintptr_t pb   = ((uintptr_t)(csr + EE) + 63) & ~(uintptr_t)63;
  __bf16* PWe  = (__bf16*)pb;                   //  6*3*512 bf16
  __bf16* PWih = PWe  + (size_t)6  * 3 * 512;   // 18*3*512 bf16
  __bf16* PWhh = PWih + (size_t)18 * 3 * 512;   // 18*3*512 bf16

  // ---- one-time per launch: pack weights + build CSR ----
  pack_w_kernel<<<6 * 3, 512, 0, stream>>>(W_e, PWe);
  pack_w_kernel<<<18 * 3, 512, 0, stream>>>(W_ih, PWih);
  pack_w_kernel<<<18 * 3, 512, 0, stream>>>(W_hh, PWhh);

  zero_int_kernel<<<NB, 256, 0, stream>>>(fill, NN);
  count_kernel<<<EE / 256, 256, 0, stream>>>(dst, fill);
  scan_block_kernel<<<NB, 256, 0, stream>>>(fill, row_start, bsums);
  scan_sums_kernel<<<1, 256, 0, stream>>>(bsums, NB);
  add_off_kernel<<<NB, 256, 0, stream>>>(row_start, bsums);
  set_last_kernel<<<1, 1, 0, stream>>>(row_start);
  zero_int_kernel<<<NB, 256, 0, stream>>>(fill, NN);
  fill_csr_kernel<<<EE / 256, 256, 0, stream>>>(src, dst, row_start, fill, csr);

  // ---- 4 propagation steps ----
  const float* h = x;                 // first step reads the input directly
  float* spare   = P;                 // buffer for m (and h_out when s > 0)

  for (int s = 0; s < 4; ++s) {
    float* m = spare;
    msg_kernel<<<ROWB, 192, 0, stream>>>(h, PWe, b_e, m);
    gather_kernel<<<NN, 96, 0, stream>>>(m, row_start, csr, Q);

    float* h_next = (s == 0) ? R : m;       // m is dead once 'a' is built
    gru_kernel<<<ROWB, 192, 0, stream>>>(Q, h, PWih, PWhh, b_ih, b_hh, h_next);

    h     = h_next;
    spare = (h == P) ? R : P;
  }

  head_kernel<<<NN / 4, 128, 0, stream>>>(h, W_fc, b_fc, (float*)d_out);
}
